// HyperbolicSageConv_56255481643036
// MI455X (gfx1250) — matrix operand order repaired and checked by reading, not compiled, not run on hardware
//
#include <hip/hip_runtime.h>
#include <hip/hip_bf16.h>
#include <math.h>

// ---------------------------------------------------------------------------
// HyperbolicSageConv on MI455X (gfx1250, wave32, WMMA, TDM).
//
//   K1 k_logmap      : u = log0(x) (bf16) + rxx = ||x_row||^2
//   K2 k_swizzle_w   : embed/layer -> WMMA B-layout bf16
//   K3 k_gemm_embed  : t = u@embed ; epilogue exp0 -> h_add(embed_bias) -> log0
//                      => g (bf16 row-major)
//   K4 k_repack_g    : g row-major -> WMMA B-layout (gsw)
//   K5 k_gemm_adj    : partial P = adj@g over a K-split (4-way, 256 WGs).
//                      adj streamed fp32->bf16 in regs; B panel staged into
//                      double-buffered LDS by the Tensor Data Mover
//                      (tensor_load_to_lds + s_wait_tensorcnt).
//   K6 k_finish      : sum partials; neigh=exp0(P); a256 = s256*[x;neigh] bf16
//   K7 k_gemm_layer  : t2 = a256@layer ; epilogue exp0 -> h_add(layer_bias)
// ---------------------------------------------------------------------------

#define NROWS  8192
#define FDIM   128
#define NSPLIT 4
#define KSPLIT (NROWS / NSPLIT)   // 2048
#define EPSV   1e-7f
#define MAXTH  (1.0f - 1e-6f)

typedef __attribute__((ext_vector_type(16))) __bf16 bf16x16;
typedef __attribute__((ext_vector_type(8)))  __bf16 bf16x8;
typedef __attribute__((ext_vector_type(4)))  __bf16 bf16x4;
typedef __attribute__((ext_vector_type(8)))  float  f32x8;
typedef __attribute__((ext_vector_type(4)))  float  f32x4;
typedef __attribute__((ext_vector_type(4)))  unsigned int u32x4;
typedef __attribute__((ext_vector_type(8)))  int    i32x8;
typedef __attribute__((ext_vector_type(4)))  int    i32x4;

// D = A(16x32 bf16) * B(32x16 bf16) + C(16x16 f32)
__device__ __forceinline__ f32x8 wmma_bf16(bf16x16 a, bf16x16 b, f32x8 c) {
  return __builtin_amdgcn_wmma_f32_16x16x32_bf16(false, a, false, b,
                                                 (short)0, c, false, false);
}

__device__ __forceinline__ float half_reduce(float v) {
  v += __shfl_xor(v, 1, 32);
  v += __shfl_xor(v, 2, 32);
  v += __shfl_xor(v, 4, 32);
  v += __shfl_xor(v, 8, 32);
  return v;
}

__device__ __forceinline__ bf16x16 load_a_bf16(const __bf16* __restrict__ base,
                                               int lda, int k0, int lane) {
  const int m  = lane & 15;
  const int kb = k0 + ((lane >> 4) << 4);
  const __bf16* p = base + (size_t)m * lda + kb;
  bf16x8 lo = *(const bf16x8*)(p);
  bf16x8 hi = *(const bf16x8*)(p + 8);
  bf16x16 a;
#pragma unroll
  for (int j = 0; j < 8; ++j) { a[j] = lo[j]; a[8 + j] = hi[j]; }
  return a;
}

__device__ __forceinline__ bf16x16 load_a_f32(const float* __restrict__ base,
                                              int lda, int k0, int lane) {
  const int m  = lane & 15;
  const int kb = k0 + ((lane >> 4) << 4);
  const float* p = base + (size_t)m * lda + kb;
  f32x4 v0 = *(const f32x4*)(p);
  f32x4 v1 = *(const f32x4*)(p + 4);
  f32x4 v2 = *(const f32x4*)(p + 8);
  f32x4 v3 = *(const f32x4*)(p + 12);
  bf16x16 a;
#pragma unroll
  for (int j = 0; j < 4; ++j) {
    a[j]      = (__bf16)v0[j];
    a[4 + j]  = (__bf16)v1[j];
    a[8 + j]  = (__bf16)v2[j];
    a[12 + j] = (__bf16)v3[j];
  }
  return a;
}

// ---- Tensor Data Mover: contiguous 1D block, global -> LDS ----------------
// D# per CDNA5 ISA ch.8: Group0 {count=1, lds_addr, global_addr, type=2},
// Group1 {data_size=8B, tensor_dim0=tile_dim0=nbytes/8, stride=nbytes/8}.
__device__ __forceinline__ void tdm_load_chunk(const void* gptr,
                                               unsigned lds_off,
                                               unsigned nbytes) {
  const unsigned long long ga = (unsigned long long)(uintptr_t)gptr;
  const unsigned elems = nbytes >> 3;  // 8-byte elements
  u32x4 g0;
  g0[0] = 1u;                                   // count=1 (valid user D#)
  g0[1] = lds_off;                              // LDS byte address
  g0[2] = (unsigned)(ga & 0xffffffffu);         // global_addr[31:0]
  g0[3] = (unsigned)(ga >> 32) | 0x80000000u;   // global_addr[56:32] | type=2
  i32x8 g1;
  g1[0] = 0x00030000;                           // data_size=3 (8B), no extras
  g1[1] = (int)((elems & 0xffffu) << 16);       // tensor_dim0[15:0]
  g1[2] = (int)((elems >> 16) | (1u << 16));    // tensor_dim0[31:16], dim1=1
  g1[3] = (int)((elems & 0xffffu) << 16);       // tile_dim0
  g1[4] = 0;                                    // tile_dim1=0, tile_dim2=0
  g1[5] = (int)elems;                           // tensor_dim0_stride[31:0]
  g1[6] = 0;
  g1[7] = 0;
  i32x4 z = {0, 0, 0, 0};
#if __clang_major__ >= 23
  i32x8 z8 = {0, 0, 0, 0, 0, 0, 0, 0};
  __builtin_amdgcn_tensor_load_to_lds(g0, g1, z, z, z8, 0);
#else
  __builtin_amdgcn_tensor_load_to_lds(g0, g1, z, z, 0);
#endif
}

// ------------------------------ K1: log0(x) --------------------------------
__global__ void __launch_bounds__(256)
k_logmap(const float* __restrict__ x, __bf16* __restrict__ u,
         float* __restrict__ rxx) {
  const int row  = blockIdx.x * 8 + (threadIdx.x >> 5);
  const int lane = threadIdx.x & 31;
  const float* xr = x + (size_t)row * FDIM;
  f32x4 v = *(const f32x4*)(xr + lane * 4);
  float ss = v[0]*v[0] + v[1]*v[1] + v[2]*v[2] + v[3]*v[3];
#pragma unroll
  for (int msk = 1; msk < 32; msk <<= 1) ss += __shfl_xor(ss, msk, 32);
  const float n = fmaxf(sqrtf(ss), EPSV);
  const float s = atanhf(fminf(n, MAXTH)) / n;
  bf16x4 o;
#pragma unroll
  for (int j = 0; j < 4; ++j) o[j] = (__bf16)(s * v[j]);
  *(bf16x4*)(u + (size_t)row * FDIM + lane * 4) = o;
  if (lane == 0) rxx[row] = ss;
}

// --------- K2: weight fp32 row-major -> WMMA B-layout bf16 -----------------
__global__ void k_swizzle_w(const float* __restrict__ w,
                            __bf16* __restrict__ wsw, int K) {
  const int rec  = blockIdx.x * blockDim.x + threadIdx.x;
  const int lane = rec & 31;
  const int t    = (rec >> 5) & 7;
  const int kb   = rec >> 8;
  if (kb >= (K >> 5)) return;
  const int n = lane & 15, kh = lane >> 4;
  const float* p = w + (size_t)(kb * 32 + kh * 16) * FDIM + t * 16 + n;
  bf16x16 v;
#pragma unroll
  for (int j = 0; j < 16; ++j) v[j] = (__bf16)p[(size_t)j * FDIM];
  *(bf16x16*)(wsw + (size_t)rec * 16) = v;
}

// --------- K4: g row-major bf16 -> WMMA B-layout (8192x128) ----------------
__global__ void k_repack_g(const __bf16* __restrict__ g,
                           __bf16* __restrict__ gsw) {
  const int rec  = blockIdx.x * blockDim.x + threadIdx.x;  // 65536 records
  const int lane = rec & 31;
  const int t    = (rec >> 5) & 7;
  const int kb   = rec >> 8;
  const int n = lane & 15, kh = lane >> 4;
  const __bf16* p = g + (size_t)(kb * 32 + kh * 16) * FDIM + t * 16 + n;
  bf16x16 v;
#pragma unroll
  for (int j = 0; j < 16; ++j) v[j] = p[(size_t)j * FDIM];
  *(bf16x16*)(gsw + (size_t)rec * 16) = v;
}

// ------- K3: t = u@embed ; g = log0(h_add(exp0(t), embed_bias)) ------------
__global__ void __launch_bounds__(256)
k_gemm_embed(const __bf16* __restrict__ u, const __bf16* __restrict__ wsw,
             const float* __restrict__ bias, __bf16* __restrict__ g) {
  const int lane  = threadIdx.x & 31;
  const int strip = (blockIdx.x * 8 + (threadIdx.x >> 5)) * 16;
  const __bf16* A = u + (size_t)strip * FDIM;
  f32x8 acc[8] = {};
#pragma unroll
  for (int ks = 0; ks < 4; ++ks) {
    bf16x16 a = load_a_bf16(A, FDIM, ks * 32, lane);
#pragma unroll
    for (int t = 0; t < 8; ++t) {
      bf16x16 b = *(const bf16x16*)(wsw + (((size_t)ks * 8 + t) * 32 + lane) * 16);
      acc[t] = wmma_bf16(a, b, acc[t]);
    }
  }
  const int n = lane & 15;
  float bv[8], bbp = 0.f;
#pragma unroll
  for (int t = 0; t < 8; ++t) { bv[t] = bias[t * 16 + n]; bbp += bv[t] * bv[t]; }
  const float bb = half_reduce(bbp);
#pragma unroll
  for (int r = 0; r < 8; ++r) {
    float s2 = 0.f, sb = 0.f;
#pragma unroll
    for (int t = 0; t < 8; ++t) { float v = acc[t][r]; s2 += v * v; sb += v * bv[t]; }
    const float nt2 = half_reduce(s2);
    const float tb  = half_reduce(sb);
    const float nt    = fmaxf(sqrtf(nt2), EPSV);
    const float th    = tanhf(nt);
    const float alpha = th / nt;
    const float x2  = alpha * alpha * nt2;
    const float xy  = alpha * tb;
    const float den = fmaxf(1.f + 2.f * xy + x2 * bb, EPSV);
    const float ca  = (1.f + 2.f * xy + bb) * alpha / den;
    const float cb  = (1.f - x2) / den;
    const float n2 = ca * ca * nt2 + 2.f * ca * cb * tb + cb * cb * bb;
    const float nn = fmaxf(sqrtf(n2), EPSV);
    const float s  = atanhf(fminf(nn, MAXTH)) / nn;
    const int m = strip + r + ((lane >> 4) << 3);
    __bf16* gr = g + (size_t)m * FDIM;
#pragma unroll
    for (int t = 0; t < 8; ++t)
      gr[t * 16 + n] = (__bf16)(s * (ca * acc[t][r] + cb * bv[t]));
  }
}

// ------- K5: partial P = adj@g over one K-split ----------------------------
#define KC_KB     4                       // K-chunk = 128
#define CHUNK_BF  (KC_KB * 8 * 32 * 16)   // 16384 bf16 = 32KB
__global__ void __launch_bounds__(256)
k_gemm_adj(const float* __restrict__ adj, const __bf16* __restrict__ gsw,
           float* __restrict__ pacc) {
  __shared__ __align__(128) __bf16 lds[2][CHUNK_BF];  // 64KB double buffer
  const int lane  = threadIdx.x & 31;
  const int wave  = threadIdx.x >> 5;
  const int strip = (blockIdx.x * 8 + wave) * 16;
  const int k0g   = blockIdx.y * KSPLIT;
  const float* A = adj + (size_t)strip * NROWS;
  const unsigned lds0 = (unsigned)(uintptr_t)(&lds[0][0]);
  f32x8 acc[8] = {};
  const int NCHUNK = KSPLIT / (KC_KB * 32);  // 16
  if (wave == 0)
    tdm_load_chunk(gsw + (size_t)(k0g >> 5) * (8 * 32 * 16), lds0, 32768);
  for (int c = 0; c < NCHUNK; ++c) {
    const int kc = k0g + c * (KC_KB * 32);
    if (wave == 0) __builtin_amdgcn_s_wait_tensorcnt(0);
    __syncthreads();  // buffer (c&1) ready for everyone; prior reads done
    if (c + 1 < NCHUNK) {
      if (wave == 0)
        tdm_load_chunk(gsw + (size_t)((kc + KC_KB * 32) >> 5) * (8 * 32 * 16),
                       lds0 + (unsigned)(((c + 1) & 1) * (CHUNK_BF * 2)),
                       32768);
      __builtin_prefetch(A + (size_t)(lane & 15) * NROWS + kc + KC_KB * 32,
                         0, 1);
    }
    const __bf16* lb = &lds[c & 1][0];
#pragma unroll
    for (int kk = 0; kk < KC_KB; ++kk) {
      bf16x16 a = load_a_f32(A, NROWS, kc + kk * 32, lane);
#pragma unroll
      for (int t = 0; t < 8; ++t) {
        bf16x16 b = *(const bf16x16*)(lb + (((size_t)kk * 8 + t) * 32 + lane) * 16);
        acc[t] = wmma_bf16(a, b, acc[t]);
      }
    }
  }
  // write fp32 partials for this split
  float* pr = pacc + ((size_t)blockIdx.y * NROWS + strip) * FDIM;
  const int n = lane & 15;
#pragma unroll
  for (int r = 0; r < 8; ++r) {
    const int m = r + ((lane >> 4) << 3);
    float* row = pr + (size_t)m * FDIM;
#pragma unroll
    for (int t = 0; t < 8; ++t) row[t * 16 + n] = acc[t][r];
  }
}

// ------- K6: combine splits; neigh=exp0(P); a256 = s256*[x;neigh] ----------
__global__ void __launch_bounds__(256)
k_finish(const float* __restrict__ pacc, const float* __restrict__ x,
         const float* __restrict__ rxx, __bf16* __restrict__ a256) {
  const int row  = blockIdx.x * 8 + (threadIdx.x >> 5);
  const int lane = threadIdx.x & 31;
  f32x4 p = {};
#pragma unroll
  for (int sp = 0; sp < NSPLIT; ++sp)
    p += *(const f32x4*)(pacc + ((size_t)sp * NROWS + row) * FDIM + lane * 4);
  float ss = p[0]*p[0] + p[1]*p[1] + p[2]*p[2] + p[3]*p[3];
#pragma unroll
  for (int msk = 1; msk < 32; msk <<= 1) ss += __shfl_xor(ss, msk, 32);
  const float np   = fmaxf(sqrtf(ss), EPSV);
  const float beta = tanhf(np) / np;              // neigh = beta * P
  const float c2   = rxx[row] + beta * beta * ss; // ||[x;neigh]||^2
  const float cn   = fmaxf(sqrtf(c2), EPSV);
  const float s    = atanhf(fminf(cn, MAXTH)) / cn;
  f32x4 xv = *(const f32x4*)(x + (size_t)row * FDIM + lane * 4);
  bf16x4 ox, on;
#pragma unroll
  for (int j = 0; j < 4; ++j) {
    ox[j] = (__bf16)(s * xv[j]);
    on[j] = (__bf16)(s * beta * p[j]);
  }
  __bf16* ar = a256 + (size_t)row * 256;
  *(bf16x4*)(ar + lane * 4)       = ox;
  *(bf16x4*)(ar + 128 + lane * 4) = on;
}

// ------- K7: out = h_add(exp0(a256 @ layer), layer_bias) -------------------
__global__ void __launch_bounds__(256)
k_gemm_layer(const __bf16* __restrict__ a256, const __bf16* __restrict__ wsw,
             const float* __restrict__ bias, float* __restrict__ out) {
  const int lane  = threadIdx.x & 31;
  const int strip = (blockIdx.x * 8 + (threadIdx.x >> 5)) * 16;
  const __bf16* A = a256 + (size_t)strip * 256;
  f32x8 acc[8] = {};
#pragma unroll
  for (int ks = 0; ks < 8; ++ks) {
    bf16x16 a = load_a_bf16(A, 256, ks * 32, lane);
#pragma unroll
    for (int t = 0; t < 8; ++t) {
      bf16x16 b = *(const bf16x16*)(wsw + (((size_t)ks * 8 + t) * 32 + lane) * 16);
      acc[t] = wmma_bf16(a, b, acc[t]);
    }
  }
  const int n = lane & 15;
  float bv[8], bbp = 0.f;
#pragma unroll
  for (int t = 0; t < 8; ++t) { bv[t] = bias[t * 16 + n]; bbp += bv[t] * bv[t]; }
  const float bb = half_reduce(bbp);
#pragma unroll
  for (int r = 0; r < 8; ++r) {
    float s2 = 0.f, sb = 0.f;
#pragma unroll
    for (int t = 0; t < 8; ++t) { float v = acc[t][r]; s2 += v * v; sb += v * bv[t]; }
    const float nt2 = half_reduce(s2);
    const float tb  = half_reduce(sb);
    const float nt    = fmaxf(sqrtf(nt2), EPSV);
    const float th    = tanhf(nt);
    const float alpha = th / nt;
    const float x2  = alpha * alpha * nt2;
    const float xy  = alpha * tb;
    const float den = fmaxf(1.f + 2.f * xy + x2 * bb, EPSV);
    const float ca  = (1.f + 2.f * xy + bb) * alpha / den;
    const float cb  = (1.f - x2) / den;
    const int m = strip + r + ((lane >> 4) << 3);
    float* orow = out + (size_t)m * FDIM;
#pragma unroll
    for (int t = 0; t < 8; ++t) orow[t * 16 + n] = ca * acc[t][r] + cb * bv[t];
  }
}

// ---------------------------------------------------------------------------
extern "C" void kernel_launch(void* const* d_in, const int* in_sizes, int n_in,
                              void* d_out, int out_size, void* d_ws,
                              size_t ws_size, hipStream_t stream) {
  const float* x     = (const float*)d_in[0];
  const float* adj   = (const float*)d_in[1];
  const float* embed = (const float*)d_in[2];
  const float* layer = (const float*)d_in[3];
  const float* ebias = (const float*)d_in[4];
  const float* lbias = (const float*)d_in[5];
  float* out = (float*)d_out;

  char* ws = (char*)d_ws;
  size_t off = 0;
  auto take = [&](size_t bytes) {
    off = (off + 255) & ~(size_t)255;
    char* p = ws + off;
    off += bytes;
    return p;
  };
  __bf16* u        = (__bf16*)take((size_t)NROWS * FDIM * 2);          // 2 MB
  float*  rxx      = (float*) take((size_t)NROWS * 4);                 // 32 KB
  __bf16* embed_sw = (__bf16*)take((size_t)FDIM * FDIM * 2);           // 32 KB
  __bf16* layer_sw = (__bf16*)take((size_t)2 * FDIM * FDIM * 2);       // 64 KB
  __bf16* g_rm     = (__bf16*)take((size_t)NROWS * FDIM * 2);          // 2 MB
  __bf16* g_sw     = (__bf16*)take((size_t)NROWS * FDIM * 2);          // 2 MB
  __bf16* a256     = (__bf16*)take((size_t)NROWS * 256 * 2);           // 4 MB
  float*  pacc     = (float*) take((size_t)NSPLIT * NROWS * FDIM * 4); // 16 MB
  (void)ws_size; (void)in_sizes; (void)n_in; (void)out_size;

  k_logmap    <<<NROWS / 8, 256, 0, stream>>>(x, u, rxx);
  k_swizzle_w <<<4, 256, 0, stream>>>(embed, embed_sw, 128);
  k_swizzle_w <<<8, 256, 0, stream>>>(layer, layer_sw, 256);
  k_gemm_embed<<<NROWS / 128, 256, 0, stream>>>(u, embed_sw, ebias, g_rm);
  k_repack_g  <<<256, 256, 0, stream>>>(g_rm, g_sw);
  k_gemm_adj  <<<dim3(NROWS / 128, NSPLIT), 256, 0, stream>>>(adj, g_sw, pacc);
  k_finish    <<<NROWS / 8, 256, 0, stream>>>(pacc, x, rxx, a256);
  k_gemm_layer<<<NROWS / 128, 256, 0, stream>>>(a256, layer_sw, lbias, out);
}
